// BertSelfAttention_51299089384027
// MI455X (gfx1250) — matrix operand order repaired and checked by reading, not compiled, not run on hardware
//
#include <hip/hip_runtime.h>
#include <math.h>

// Problem dims (fixed by the reference)
#define B_   4
#define S_   2048
#define H_   1024
#define NH_  16
#define HD_  64
#define M_   (B_ * S_)   // 8192 rows of X

typedef __attribute__((ext_vector_type(16))) __bf16 bf16x16;
typedef __attribute__((ext_vector_type(8)))  __bf16 bf16x8;
typedef __attribute__((ext_vector_type(8)))  float  f32x8;
typedef __attribute__((ext_vector_type(4)))  int    v4i;

#define AS1 __attribute__((address_space(1)))
#define AS3 __attribute__((address_space(3)))

#if __has_builtin(__builtin_amdgcn_global_load_async_to_lds_b128)
#define HAS_ASYNC 1
#else
#define HAS_ASYNC 0
#endif

// One 16-byte async (ASYNCcnt-tracked) global->LDS copy per lane.
__device__ __forceinline__ void async_copy16(const __bf16* g, __bf16* l) {
#if HAS_ASYNC
  __builtin_amdgcn_global_load_async_to_lds_b128(
      (AS1 v4i*)(g), (AS3 v4i*)(l), 0, 0);
#else
  *(bf16x8*)l = *(const bf16x8*)g;   // synchronous fallback
#endif
}

__device__ __forceinline__ void wait_async() {
#if HAS_ASYNC
#if __has_builtin(__builtin_amdgcn_s_wait_asynccnt)
  __builtin_amdgcn_s_wait_asynccnt(0);
#else
  asm volatile("s_wait_asynccnt 0" ::: "memory");
#endif
#endif
}

// ---------------------------------------------------------------------------
// fp32 -> bf16 conversion (X and weights)
// ---------------------------------------------------------------------------
__global__ void cvt_f32_bf16(const float* __restrict__ src,
                             __bf16* __restrict__ dst, int n) {
  int i = blockIdx.x * blockDim.x + threadIdx.x;
  if (i < n) dst[i] = (__bf16)src[i];
}

// ---------------------------------------------------------------------------
// WMMA fragment loaders (wave32, 16x16x32 bf16 shapes)
//
// A matrix 16x32 (MxK), 16-bit, ISA 7.12.2:
//   lane L: row M = L&15 ; halves 0..7 = K kb+ko..+7, halves 8..15 = K kb+16+ko..+7
//   with ko = (L>>4)*8.  => two contiguous b128 loads from a K-contiguous row.
//
// B matrix 32x16 (KxN), 16-bit: lane L holds column N = L&15, halves 0..7 =
//   K kb+ko..+7, halves 8..15 = K kb+16+ko..+7, ko = (L>>4)*8 (same as A).
// ---------------------------------------------------------------------------
__device__ __forceinline__ bf16x16 load_frag_a(const __bf16* base, int ld, int kb) {
  int lane = threadIdx.x & 31;
  int r    = lane & 15;
  int ko   = (lane >> 4) * 8;
  const __bf16* p = base + (size_t)r * ld + kb + ko;
  bf16x8 lo = *(const bf16x8*)(p);
  bf16x8 hi = *(const bf16x8*)(p + 16);
  bf16x16 f;
#pragma unroll
  for (int i = 0; i < 8; i++) { f[i] = lo[i]; f[i + 8] = hi[i]; }
  return f;
}

__device__ __forceinline__ bf16x16 load_frag_b(const __bf16* base, int ld, int kb) {
  int lane = threadIdx.x & 31;
  int c    = lane & 15;
  int ko   = (lane >> 4) * 8;
  const __bf16* p = base + (size_t)c * ld + kb + ko;
  bf16x8 lo = *(const bf16x8*)(p);
  bf16x8 hi = *(const bf16x8*)(p + 16);
  bf16x16 f;
#pragma unroll
  for (int i = 0; i < 8; i++) { f[i] = lo[i]; f[i + 8] = hi[i]; }
  return f;
}

// ---------------------------------------------------------------------------
// Phase 1: QKV projection GEMM.  D = X * W^T + bias  (all K-contiguous).
// grid = (N/64, M/128, 3) ; block = 256 (8 waves).  Wave w owns a 16-row
// strip and a 64-wide N tile (4 fp32 accumulators), K loop of 32.
// z==0 -> Q [B,NH,S,HD], z==1 -> K [B,NH,S,HD], z==2 -> V^T [B,NH,HD,S].
// ---------------------------------------------------------------------------
__global__ void __launch_bounds__(256) qkv_gemm(
    const __bf16* __restrict__ X,
    const __bf16* __restrict__ Wqb, const __bf16* __restrict__ Wkb,
    const __bf16* __restrict__ Wvb,
    const float* __restrict__ bq, const float* __restrict__ bk,
    const float* __restrict__ bv,
    __bf16* __restrict__ Qo, __bf16* __restrict__ Ko, __bf16* __restrict__ VTo) {
  const int z = blockIdx.z;
  const __bf16* W   = (z == 0) ? Wqb : (z == 1) ? Wkb : Wvb;
  const float* bias = (z == 0) ? bq  : (z == 1) ? bk  : bv;

  const int wave = threadIdx.x >> 5;
  const int lane = threadIdx.x & 31;
  const int m0 = blockIdx.y * 128 + wave * 16;
  const int n0 = blockIdx.x * 64;

  const __bf16* Arow = X + (size_t)m0 * H_;

  f32x8 acc[4] = {};
  for (int k = 0; k < H_; k += 32) {
    // near-cache speculative prefetch of the next A chunk
    __builtin_prefetch(Arow + (size_t)(lane & 15) * H_ + k + 64, 0, 3);
    bf16x16 a = load_frag_a(Arow, H_, k);
#pragma unroll
    for (int n = 0; n < 4; n++) {
      bf16x16 b = load_frag_b(W + (size_t)(n0 + n * 16) * H_, H_, k);
      acc[n] = __builtin_amdgcn_wmma_f32_16x16x32_bf16(
          false, a, false, b, (short)0, acc[n], false, false);
    }
  }

  // Epilogue: bias add + bf16 store in attention-friendly layouts.
  const int bb    = m0 / S_;
  const int sbase = m0 % S_;
  const int rlo   = (lane >> 4) * 8;
  const int col0  = n0 + (lane & 15);
#pragma unroll
  for (int n = 0; n < 4; n++) {
    const int col = col0 + n * 16;
    const int hh  = col >> 6;   // head
    const int dd  = col & 63;   // dim within head
    const float bval = bias[col];
#pragma unroll
    for (int i = 0; i < 8; i++) {
      const int s = sbase + rlo + i;
      const float v = acc[n][i] + bval;
      if (z == 2) {
        VTo[(((size_t)bb * NH_ + hh) * HD_ + dd) * S_ + s] = (__bf16)v;
      } else {
        __bf16* dst = (z == 0) ? Qo : Ko;
        dst[(((size_t)bb * NH_ + hh) * S_ + s) * HD_ + dd] = (__bf16)v;
      }
    }
  }
}

// ---------------------------------------------------------------------------
// Phase 2: flash attention.  grid = (S/64, B*NH) ; block = 128 (4 waves).
// K/V tiles (8 KB each) are staged into LDS with double-buffered
// GLOBAL_LOAD_ASYNC_TO_LDS_B128 (ASYNCcnt) so all 4 waves share one copy and
// the next tile's DMA overlaps the current tile's WMMA work.
// Each wave owns 16 query rows with online softmax; C-layout rows live in
// 16-lane halves -> row reductions via __shfl_xor {1,2,4,8}.  P is
// re-swizzled C-layout -> A-layout through a per-wave LDS tile.
// ---------------------------------------------------------------------------

// Stage one 64x64 bf16 K tile and one 64x64 bf16 V^T tile into LDS.
// 128 threads * 4 chunks * 16 B = 8 KB per tile.
__device__ __forceinline__ void issue_tile(const __bf16* __restrict__ Kh,
                                           const __bf16* __restrict__ Vh,
                                           __bf16* kl, __bf16* vl, int t) {
  const int tid = threadIdx.x;   // 0..127
#pragma unroll
  for (int u = 0; u < 4; u++) {
    const int idx = tid + u * 128;   // chunk 0..511
    const int row = idx >> 3;        // 0..63
    const int ch  = (idx & 7) * 8;   // bf16 offset within row
    async_copy16(Kh + (size_t)(t + row) * HD_ + ch, kl + row * 64 + ch);
    async_copy16(Vh + (size_t)row * S_ + t + ch,    vl + row * 64 + ch);
  }
}

__global__ void __launch_bounds__(128) flash_attn(
    const __bf16* __restrict__ Q, const __bf16* __restrict__ Kh_,
    const __bf16* __restrict__ VT, const float* __restrict__ mask,
    float* __restrict__ out) {
  __shared__ alignas(16) __bf16 Klds[2][64 * 64];
  __shared__ alignas(16) __bf16 Vlds[2][64 * 64];
  __shared__ alignas(16) __bf16 Plds[4][16 * 64];

  const int wave = threadIdx.x >> 5;
  const int lane = threadIdx.x & 31;
  const int bh = blockIdx.y;
  const int b  = bh / NH_;
  const int h  = bh % NH_;
  const int q0 = blockIdx.x * 64 + wave * 16;

  const __bf16* Qh = Q   + (size_t)bh * S_ * HD_;
  const __bf16* Kh = Kh_ + (size_t)bh * S_ * HD_;
  const __bf16* Vh = VT  + (size_t)bh * HD_ * S_;
  const float*  mk = mask + (size_t)b * S_;

  // Q fragments are reused for every key tile.
  const bf16x16 qa0 = load_frag_a(Qh + (size_t)q0 * HD_, HD_, 0);
  const bf16x16 qa1 = load_frag_a(Qh + (size_t)q0 * HD_, HD_, 32);

  f32x8 acc[4] = {};
  float mrow[8], lrow[8];
#pragma unroll
  for (int i = 0; i < 8; i++) { mrow[i] = -1e30f; lrow[i] = 0.0f; }

  const float scale = 0.125f;  // 1/sqrt(HD)
  __bf16* pl = &Plds[wave][0];

  // Prime the pipeline: async-stage tile 0 into buffer 0.
  issue_tile(Kh, Vh, Klds[0], Vlds[0], 0);
  int buf = 0;

  for (int t = 0; t < S_; t += 64) {
    // Own async copies done -> barrier publishes the tile to all waves.
    wait_async();
    __syncthreads();
    // Prefetch the next tile into the other buffer (safe: the barrier above
    // proves every wave finished computing on it last iteration).
    if (t + 64 < S_) issue_tile(Kh, Vh, Klds[buf ^ 1], Vlds[buf ^ 1], t + 64);

    const __bf16* Kt = Klds[buf];
    const __bf16* Vt = Vlds[buf];

    // ---- scores: 16(q) x 64(k) in four 16x16 fp32 tiles ----
    f32x8 st[4];
#pragma unroll
    for (int n = 0; n < 4; n++) {
      bf16x16 kb0 = load_frag_b(Kt + (n * 16) * 64, 64, 0);
      bf16x16 kb1 = load_frag_b(Kt + (n * 16) * 64, 64, 32);
      f32x8 s = {};
      s = __builtin_amdgcn_wmma_f32_16x16x32_bf16(false, qa0, false, kb0,
                                                  (short)0, s, false, false);
      s = __builtin_amdgcn_wmma_f32_16x16x32_bf16(false, qa1, false, kb1,
                                                  (short)0, s, false, false);
      const float mv = mk[t + n * 16 + (lane & 15)];
#pragma unroll
      for (int i = 0; i < 8; i++) s[i] = s[i] * scale + mv;
      st[n] = s;
    }

    // ---- online softmax: new row max ----
    float nm[8];
#pragma unroll
    for (int i = 0; i < 8; i++) {
      nm[i] = mrow[i];
#pragma unroll
      for (int n = 0; n < 4; n++) nm[i] = fmaxf(nm[i], st[n][i]);
    }
#pragma unroll
    for (int i = 0; i < 8; i++) {
      nm[i] = fmaxf(nm[i], __shfl_xor(nm[i], 1, 32));
      nm[i] = fmaxf(nm[i], __shfl_xor(nm[i], 2, 32));
      nm[i] = fmaxf(nm[i], __shfl_xor(nm[i], 4, 32));
      nm[i] = fmaxf(nm[i], __shfl_xor(nm[i], 8, 32));
    }

    float corr[8], rs[8];
#pragma unroll
    for (int i = 0; i < 8; i++) {
      corr[i] = __expf(mrow[i] - nm[i]);
      mrow[i] = nm[i];
      rs[i]   = 0.0f;
    }

    // ---- P = exp(S - m); stage to per-wave LDS tile (q,k row-major) ----
    const int rhalf = (lane >> 4) * 8;
    const int cl    = lane & 15;
#pragma unroll
    for (int n = 0; n < 4; n++) {
#pragma unroll
      for (int i = 0; i < 8; i++) {
        const float p = __expf(st[n][i] - nm[i]);
        rs[i] += p;
        pl[(rhalf + i) * 64 + n * 16 + cl] = (__bf16)p;
      }
    }

    // row sums + running denominator / accumulator rescale
#pragma unroll
    for (int i = 0; i < 8; i++) {
      rs[i] += __shfl_xor(rs[i], 1, 32);
      rs[i] += __shfl_xor(rs[i], 2, 32);
      rs[i] += __shfl_xor(rs[i], 4, 32);
      rs[i] += __shfl_xor(rs[i], 8, 32);
      lrow[i] = lrow[i] * corr[i] + rs[i];
    }
#pragma unroll
    for (int nd = 0; nd < 4; nd++)
#pragma unroll
      for (int i = 0; i < 8; i++) acc[nd][i] *= corr[i];

    // ---- P · V : A from per-wave LDS (A-layout), B from LDS V^T tile ----
    bf16x16 pa0 = load_frag_a(pl, 64, 0);
    bf16x16 pa1 = load_frag_a(pl, 64, 32);
#pragma unroll
    for (int nd = 0; nd < 4; nd++) {
      bf16x16 vb0 = load_frag_b(Vt + (nd * 16) * 64, 64, 0);
      bf16x16 vb1 = load_frag_b(Vt + (nd * 16) * 64, 64, 32);
      acc[nd] = __builtin_amdgcn_wmma_f32_16x16x32_bf16(
          false, pa0, false, vb0, (short)0, acc[nd], false, false);
      acc[nd] = __builtin_amdgcn_wmma_f32_16x16x32_bf16(
          false, pa1, false, vb1, (short)0, acc[nd], false, false);
    }
    buf ^= 1;
  }

  // ---- normalize and write [B,S,H] fp32 output ----
  const int rhalf = (lane >> 4) * 8;
  const int cl    = lane & 15;
#pragma unroll
  for (int nd = 0; nd < 4; nd++) {
#pragma unroll
    for (int i = 0; i < 8; i++) {
      const int s = q0 + rhalf + i;
      const int d = nd * 16 + cl;
      out[((size_t)b * S_ + s) * H_ + h * HD_ + d] = acc[nd][i] / lrow[i];
    }
  }
}

// ---------------------------------------------------------------------------
// Host launcher.  Workspace layout (bf16 staging, ~70 MiB total):
//   Xb 16MiB | Wqb 2MiB | Wkb 2MiB | Wvb 2MiB | Q 16MiB | K 16MiB | VT 16MiB
// ---------------------------------------------------------------------------
extern "C" void kernel_launch(void* const* d_in, const int* in_sizes, int n_in,
                              void* d_out, int out_size, void* d_ws,
                              size_t ws_size, hipStream_t stream) {
  const float* hs   = (const float*)d_in[0];
  const float* mask = (const float*)d_in[1];
  const float* Wq   = (const float*)d_in[2];
  const float* bq   = (const float*)d_in[3];
  const float* Wk   = (const float*)d_in[4];
  const float* bk   = (const float*)d_in[5];
  const float* Wv   = (const float*)d_in[6];
  const float* bv   = (const float*)d_in[7];
  float* out = (float*)d_out;

  char* ws = (char*)d_ws;
  const size_t nX = (size_t)M_ * H_;   // 8,388,608
  const size_t nW = (size_t)H_ * H_;   // 1,048,576

  __bf16* Xb  = (__bf16*)ws;                  ws += nX * sizeof(__bf16);
  __bf16* Wqb = (__bf16*)ws;                  ws += nW * sizeof(__bf16);
  __bf16* Wkb = (__bf16*)ws;                  ws += nW * sizeof(__bf16);
  __bf16* Wvb = (__bf16*)ws;                  ws += nW * sizeof(__bf16);
  __bf16* Qb  = (__bf16*)ws;                  ws += nX * sizeof(__bf16);
  __bf16* Kb  = (__bf16*)ws;                  ws += nX * sizeof(__bf16);
  __bf16* VTb = (__bf16*)ws;                  ws += nX * sizeof(__bf16);

  // Phase 0: downconvert to bf16
  cvt_f32_bf16<<<(int)((nX + 255) / 256), 256, 0, stream>>>(hs, Xb, (int)nX);
  cvt_f32_bf16<<<(int)((nW + 255) / 256), 256, 0, stream>>>(Wq, Wqb, (int)nW);
  cvt_f32_bf16<<<(int)((nW + 255) / 256), 256, 0, stream>>>(Wk, Wkb, (int)nW);
  cvt_f32_bf16<<<(int)((nW + 255) / 256), 256, 0, stream>>>(Wv, Wvb, (int)nW);

  // Phase 1: QKV projections (WMMA bf16)
  qkv_gemm<<<dim3(H_ / 64, M_ / 128, 3), 256, 0, stream>>>(
      Xb, Wqb, Wkb, Wvb, bq, bk, bv, Qb, Kb, VTb);

  // Phase 2: flash attention (WMMA bf16, async LDS staging, online softmax)
  flash_attn<<<dim3(S_ / 64, B_ * NH_), 128, 0, stream>>>(Qb, Kb, VTb, mask,
                                                          out);
}
